// StyleDownBlock_47467978556050
// MI455X (gfx1250) — matrix operand order repaired and checked by reading, not compiled, not run on hardware
//
#include <hip/hip_runtime.h>
#include <cstdint>
#include <cstddef>

// ---------------------------------------------------------------------------
// StyleGAN2-style modulated 3x3 stride-2 conv block for MI455X (gfx1250).
// styles folded into activations, dcoef into epilogue => big conv is an
// implicit GEMM with ONE shared bf16 weight matrix:
//   per sample n:  Y[512,1024] = Wflip[512, 9*256] x patches(xm_n)[9*256, 1024]
// K ordered (tap, channel), xm stored NHWC bf16 so each K-step of 32 is one
// fixed tap x 32 contiguous channels => pure b128 traffic, no gather ALU.
// Core op: V_WMMA_F32_16X16X32_BF16 (f32 accumulate).
// Double-buffered LDS + GLOBAL_LOAD_ASYNC_TO_LDS_B128 staging (ASYNCcnt
// pipelined one step ahead), sync fallback if the builtin is absent.
// ---------------------------------------------------------------------------

typedef __bf16 bf16_t;
typedef __attribute__((ext_vector_type(16))) __bf16 bf16x16;
typedef __attribute__((ext_vector_type(8)))  float   floatx8;
typedef __attribute__((ext_vector_type(4)))  int     v4i_;

union Frag { bf16x16 v; uint4 q[2]; };

#define N_    16
#define CI    256
#define CO    512
#define H_    64
#define WDIM  512
#define FH    65            // filtered spatial size (H+1)
#define P_    1024          // 32*32 output pixels
#define KTOT  2304          // 9 taps * 256 channels
#define NSTEPS 72           // KTOT / 32

#if defined(__AMDGCN__) && __has_builtin(__builtin_amdgcn_global_load_async_to_lds_b128)
#define HAVE_ASYNC 1
#else
#define HAVE_ASYNC 0
#endif

typedef __attribute__((address_space(1))) void* gas_ptr;
typedef __attribute__((address_space(3))) void* lds_ptr;
typedef __attribute__((address_space(1))) v4i_* g128_t;   // builtin param 0 type
typedef __attribute__((address_space(3))) v4i_* l128_t;   // builtin param 1 type

#define ASYNC_B128(gsrc, ldst) \
    __builtin_amdgcn_global_load_async_to_lds_b128( \
        (g128_t)(gas_ptr)(gsrc), (l128_t)(lds_ptr)(ldst), 0, 0)

template <int N> __device__ __forceinline__ void wait_async() {
#if defined(__AMDGCN__) && __has_builtin(__builtin_amdgcn_s_wait_asynccnt)
    __builtin_amdgcn_s_wait_asynccnt(N);
#else
    asm volatile("s_wait_asynccnt %0" :: "n"(N) : "memory");
#endif
}

// workspace layout (bytes), all 16B aligned
static constexpr size_t OFF_STYLES = 0;                                   // 16*256 f32
static constexpr size_t OFF_DCOEF  = OFF_STYLES + (size_t)N_*CI*4;        // 16*512 f32
static constexpr size_t OFF_WSQ    = OFF_DCOEF  + (size_t)N_*CO*4;        // 512*256 f32
static constexpr size_t OFF_WFLIP  = OFF_WSQ    + (size_t)CO*CI*4;        // 512*2304 bf16
static constexpr size_t OFF_XM     = OFF_WFLIP  + (size_t)CO*KTOT*2;      // 16*65*65*256 bf16

// ---------------------------------------------------------------------------
// styles[n,i] = (w[n,:] . affine_w[i,:]) / sqrt(512) + affine_b[i]
__global__ __launch_bounds__(256) void k_styles(
    const float* __restrict__ w, const float* __restrict__ aw,
    const float* __restrict__ ab, float* __restrict__ styles)
{
    const int n = blockIdx.x, i = threadIdx.x;
    const float* wr = w  + (size_t)n * WDIM;
    const float* ar = aw + (size_t)i * WDIM;
    float acc = 0.f;
    for (int k = 0; k < WDIM; ++k) acc = fmaf(wr[k], ar[k], acc);
    styles[n*CI + i] = fmaf(acc, 0.044194173824159216f, ab[i]); // 1/sqrt(512)
}

// ---------------------------------------------------------------------------
// wsq[o,i] = sum_t weight[o,i,t]^2 ; Wflip[o][tap*256+i] = weight[o,i,8-tap]
__global__ __launch_bounds__(256) void k_wsq_wflip(
    const float* __restrict__ weight, float* __restrict__ wsq,
    bf16_t* __restrict__ Wf)
{
    const int o = blockIdx.x, i = threadIdx.x;
    const float* wr = weight + (size_t)(o*CI + i) * 9;
    float s = 0.f;
    #pragma unroll
    for (int t = 0; t < 9; ++t) {
        const float v = wr[t];
        s = fmaf(v, v, s);
        Wf[(size_t)o*KTOT + (size_t)(8 - t)*CI + i] = (bf16_t)v;
    }
    wsq[o*CI + i] = s;
}

// ---------------------------------------------------------------------------
// dcoef[n,o] = rsqrt( sum_i styles[n,i]^2 * wsq[o,i] + 1e-8 )
__global__ __launch_bounds__(256) void k_dcoef(
    const float* __restrict__ styles, const float* __restrict__ wsq,
    float* __restrict__ dcoef)
{
    const int idx = blockIdx.x * 256 + threadIdx.x;   // 8192 = 16*512
    const int n = idx >> 9, o = idx & 511;
    const float* st = styles + n*CI;
    const float* wq = wsq    + o*CI;
    float acc = 1e-8f;
    for (int i = 0; i < CI; ++i) acc = fmaf(st[i]*st[i], wq[i], acc);
    dcoef[idx] = rsqrtf(acc);
}

// ---------------------------------------------------------------------------
// Low-pass filter (separable [1,3,3,1]/8, pad 2) + modulate by styles[n,i],
// store bf16 to xm in channels-last layout [n][y][x][i].
__global__ __launch_bounds__(256) void k_filter_mod(
    const float* __restrict__ x, const float* __restrict__ styles,
    bf16_t* __restrict__ Xm)
{
    __shared__ float pad[68 * 68];
    __shared__ float hrow[68 * 68];
    const int i = blockIdx.x;       // channel
    const int n = blockIdx.y;       // sample
    const int tid = threadIdx.x;
    const float* xp = x + (size_t)(n*CI + i) * H_ * H_;

    for (int idx = tid; idx < 68*68; idx += 256) {
        const int py = idx / 68, px = idx - py*68;
        const int sy = py - 2,   sx = px - 2;
        float v = 0.f;
        if ((unsigned)sy < (unsigned)H_ && (unsigned)sx < (unsigned)H_)
            v = xp[sy*H_ + sx];
        pad[idx] = v;
    }
    __syncthreads();
    for (int idx = tid; idx < 68*FH; idx += 256) {
        const int py = idx / FH, ox = idx - py*FH;
        const float* r = &pad[py*68 + ox];
        hrow[py*68 + ox] = 0.125f*(r[0] + r[3]) + 0.375f*(r[1] + r[2]);
    }
    __syncthreads();
    const float st = styles[n*CI + i];
    for (int idx = tid; idx < FH*FH; idx += 256) {
        const int oy = idx / FH, ox = idx - oy*FH;
        const float* c = &hrow[oy*68 + ox];
        const float v = 0.125f*(c[0] + c[3*68]) + 0.375f*(c[68] + c[2*68]);
        Xm[(((size_t)n*FH + oy)*FH + ox)*CI + i] = (bf16_t)(v * st);
    }
}

// ---------------------------------------------------------------------------
// Main implicit GEMM with V_WMMA_F32_16X16X32_BF16.
// Block tile 128(M=O) x 128(N=pixels); 8 waves of 32x64; 72 K-steps of 32.
// Double-buffered LDS, async global->LDS staging pipelined one step ahead.
__global__ __launch_bounds__(256) void k_gemm(
    const bf16_t* __restrict__ Wf, const bf16_t* __restrict__ Xm,
    const float* __restrict__ dcoef, const float* __restrict__ bias,
    float* __restrict__ out)
{
    constexpr int LDA = 40;                 // bf16 row pitch (80B, 16B aligned)
    __shared__ bf16_t As[2][128 * LDA];     // 2 x 10KB
    __shared__ bf16_t Bs[2][128 * LDA];     // 2 x 10KB

    const int tid  = threadIdx.x;
    const int lane = tid & 31;
    const int wave = tid >> 5;
    const int wm   = wave & 3;              // M sub-tile (x32)
    const int wn   = wave >> 2;             // N sub-tile (x64)

    const int pblk = blockIdx.x;            // 0..7  (pixel tile)
    const int mblk = blockIdx.y;            // 0..3  (output-channel tile)
    const int n    = blockIdx.z;            // sample

    // Staging decode: each thread owns two 16B chunks per tile (rows r0, r0+64).
    const int q   = tid & 3;                // 16B chunk within 64B row
    const int r0  = tid >> 2;
    const int pg0 = pblk*128 + r0;
    const int pg1 = pg0 + 64;
    const int oh0 = pg0 >> 5, ow0 = pg0 & 31;
    const int oh1 = pg1 >> 5, ow1 = pg1 & 31;

    const bf16_t* WfA = Wf + (size_t)(mblk*128) * KTOT;

    auto stage = [&](int s, int buf) {
        const int tap = s >> 3;             // 0..8
        const int ib  = s & 7;              // channel block
        const int kh = tap / 3, kw = tap - kh*3;
        const bf16_t* a0 = WfA + (size_t)r0*KTOT + tap*256 + ib*32 + q*8;
        const bf16_t* b0 = Xm + (((size_t)n*FH + (2*oh0 + kh))*FH + (2*ow0 + kw))*CI + ib*32 + q*8;
        const bf16_t* b1 = Xm + (((size_t)n*FH + (2*oh1 + kh))*FH + (2*ow1 + kw))*CI + ib*32 + q*8;
        bf16_t* sA = &As[buf][r0*LDA + q*8];
        bf16_t* sB = &Bs[buf][r0*LDA + q*8];
        __builtin_prefetch(a0 + 32, 0, 1);  // global_prefetch next K chunk
#if HAVE_ASYNC
        ASYNC_B128(a0,            sA);
        ASYNC_B128(a0 + 64*KTOT,  sA + 64*LDA);
        ASYNC_B128(b0,            sB);
        ASYNC_B128(b1,            sB + 64*LDA);
#else
        *(uint4*)sA            = *(const uint4*)a0;
        *(uint4*)(sA + 64*LDA) = *(const uint4*)(a0 + (size_t)64*KTOT);
        *(uint4*)sB            = *(const uint4*)b0;
        *(uint4*)(sB + 64*LDA) = *(const uint4*)b1;
#endif
    };

    // Fragment addressing (ISA 7.12.2 layouts):
    // A: lane<16 holds K {0..7,16..23}; lane>=16 holds K {8..15,24..31}.
    // B: lane l = column (l&15); lane<16 holds K {0..7,16..23}; lane>=16 holds K {8..15,24..31}.
    const int arow0 = (wm*32 + (lane & 15)) * LDA;
    const int brow0 = (wn*64 + (lane & 15)) * LDA;
    const int akoff = (lane >> 4) * 8;
    const int bkoff = (lane >> 4) * 8;

    floatx8 acc[2][4] = {};

    stage(0, 0);
    for (int s = 0; s < NSTEPS; ++s) {
        const int buf = s & 1;
        if (s + 1 < NSTEPS) {
            stage(s + 1, buf ^ 1);
#if HAVE_ASYNC
            wait_async<4>();                // stage s drained (in-order), s+1 in flight
#endif
        } else {
#if HAVE_ASYNC
            wait_async<0>();
#endif
        }
        __syncthreads();                    // stage s visible to all waves

        const bf16_t* Ab = As[buf];
        const bf16_t* Bb = Bs[buf];
        Frag a[2], b[4];
        #pragma unroll
        for (int mt = 0; mt < 2; ++mt) {
            const int base = arow0 + mt*16*LDA;
            a[mt].q[0] = *(const uint4*)&Ab[base + akoff];
            a[mt].q[1] = *(const uint4*)&Ab[base + akoff + 16];
        }
        #pragma unroll
        for (int nt = 0; nt < 4; ++nt) {
            const int base = brow0 + nt*16*LDA;
            b[nt].q[0] = *(const uint4*)&Bb[base + bkoff];
            b[nt].q[1] = *(const uint4*)&Bb[base + bkoff + 16];
        }
        #pragma unroll
        for (int mt = 0; mt < 2; ++mt)
            #pragma unroll
            for (int nt = 0; nt < 4; ++nt)
                acc[mt][nt] = __builtin_amdgcn_wmma_f32_16x16x32_bf16(
                    false, a[mt].v, false, b[nt].v,
                    (short)0, acc[mt][nt], false, false);
        __syncthreads();                    // all waves done reading buf before refill
    }

    // Epilogue: demod + bias + lrelu * sqrt(2)
    const float* dcn  = dcoef + n*CO;
    float*       outn = out   + (size_t)n * CO * P_;
    #pragma unroll
    for (int mt = 0; mt < 2; ++mt) {
        const int obase = mblk*128 + wm*32 + mt*16 + ((lane >> 4) << 3);
        #pragma unroll
        for (int nt = 0; nt < 4; ++nt) {
            const int p = pblk*128 + wn*64 + nt*16 + (lane & 15);
            #pragma unroll
            for (int r = 0; r < 8; ++r) {
                const int o = obase + r;
                float y = fmaf(acc[mt][nt][r], dcn[o], bias[o]);
                y = (y >= 0.f ? y : 0.2f * y) * 1.4142135623730951f;
                outn[(size_t)o * P_ + p] = y;
            }
        }
    }
}

// ---------------------------------------------------------------------------
extern "C" void kernel_launch(void* const* d_in, const int* in_sizes, int n_in,
                              void* d_out, int out_size, void* d_ws, size_t ws_size,
                              hipStream_t stream)
{
    (void)in_sizes; (void)n_in; (void)out_size; (void)ws_size;
    const float* x  = (const float*)d_in[0];   // [16,256,64,64]
    const float* w  = (const float*)d_in[1];   // [16,512]
    const float* aw = (const float*)d_in[2];   // [256,512]
    const float* ab = (const float*)d_in[3];   // [256]
    const float* wt = (const float*)d_in[4];   // [512,256,3,3]
    const float* bs = (const float*)d_in[5];   // [512]
    float* out = (float*)d_out;                // [16,512,32,32]

    char* ws = (char*)d_ws;
    float*  styles = (float*)(ws + OFF_STYLES);
    float*  dcoef  = (float*)(ws + OFF_DCOEF);
    float*  wsq    = (float*)(ws + OFF_WSQ);
    bf16_t* Wf     = (bf16_t*)(ws + OFF_WFLIP);
    bf16_t* Xm     = (bf16_t*)(ws + OFF_XM);

    k_styles    <<<N_,             256, 0, stream>>>(w, aw, ab, styles);
    k_wsq_wflip <<<CO,             256, 0, stream>>>(wt, wsq, Wf);
    k_dcoef     <<<(N_*CO)/256,    256, 0, stream>>>(styles, wsq, dcoef);
    k_filter_mod<<<dim3(CI, N_),   256, 0, stream>>>(x, styles, Xm);
    k_gemm      <<<dim3(8, 4, N_), 256, 0, stream>>>(Wf, Xm, dcoef, bs, out);
}
